// TimeAwareAttention_9818295239587
// MI455X (gfx1250) — matrix operand order repaired and checked by reading, not compiled, not run on hardware
//
#include <hip/hip_runtime.h>

// ---------------------------------------------------------------------------
// MI455X (gfx1250) time-aware attention block, bf16 WMMA everywhere.
// B=8, S=1024, d=1024, H=16, hd=64.
// Register-blocked WMMA GEMMs: 2x2 tiles (qkv), 1x4 tiles (out-proj).
// ---------------------------------------------------------------------------

typedef __attribute__((ext_vector_type(16))) __bf16 v16bf;
typedef __attribute__((ext_vector_type(8)))  float  v8f;
typedef __attribute__((ext_vector_type(2)))  __bf16 bf16x2;

#define D_MODEL 1024
#define SEQ     1024
#define BATCH   8
#define HEADS   16
#define HDIM    64

// ---- WMMA helper -----------------------------------------------------------
__device__ __forceinline__ v8f wmma_bf16(v16bf a, v16bf b, v8f c) {
  return __builtin_amdgcn_wmma_f32_16x16x32_bf16(
      /*neg_a=*/false, a, /*neg_b=*/false, b,
      /*c_mod=*/(short)0, c, /*reuse_a=*/false, /*reuse_b=*/false);
}

// A-matrix 16x32 bf16 fragment, row-major A with leading dim lda (elements).
// Layout per ISA 7.12.2: lanes 0-15 row M=lane hold K=0..7 (VGPR0-3) and
// K=16..23 (VGPR4-7); lanes 16-31 hold K=8..15 and K=24..31.
__device__ __forceinline__ v16bf load_a_frag(const __bf16* A, int lda, int lane) {
  int mrow = lane & 15;
  int half = lane >> 4;
  const __bf16* r = A + mrow * lda + half * 8;
  v16bf a;
#pragma unroll
  for (int v = 0; v < 4; ++v) {
    bf16x2 p0 = *(const bf16x2*)(r + 2 * v);
    bf16x2 p1 = *(const bf16x2*)(r + 16 + 2 * v);
    a[2 * v]     = p0[0];
    a[2 * v + 1] = p0[1];
    a[8 + 2 * v] = p1[0];
    a[9 + 2 * v] = p1[1];
  }
  return a;
}

// B-matrix 32x16 bf16 fragment from Bt = B transposed (N x K row-major, so
// each lane reads a contiguous K-pair). Lanes 0-15: N=lane, K=0..7 and K=16..23;
// lanes 16-31: N=lane-16, K=8..15 and K=24..31 (2 values per VGPR).
__device__ __forceinline__ v16bf load_b_frag(const __bf16* Bt, int ldbt, int lane) {
  int n    = lane & 15;
  int koff = (lane >> 4) * 8;
  const __bf16* r = Bt + n * ldbt + koff;
  v16bf b;
#pragma unroll
  for (int v = 0; v < 4; ++v) {
    bf16x2 p0 = *(const bf16x2*)(r + 2 * v);
    bf16x2 p1 = *(const bf16x2*)(r + 16 + 2 * v);
    b[2 * v]     = p0[0];
    b[2 * v + 1] = p0[1];
    b[8 + 2 * v] = p1[0];
    b[9 + 2 * v] = p1[1];
  }
  return b;
}

// ---- elementwise helpers ---------------------------------------------------
__global__ __launch_bounds__(256) void cvt_f32_to_bf16(const float* __restrict__ in,
                                                       __bf16* __restrict__ out, int n) {
  int i = blockIdx.x * 256 + threadIdx.x;
  if (i < n) out[i] = (__bf16)in[i];
}

__global__ __launch_bounds__(256) void zero_f32(float* __restrict__ p, int n) {
  int i = blockIdx.x * 256 + threadIdx.x;
  if (i < n) p[i] = 0.0f;
}

// ---- qkv tile store: scatter one 16x16 f32 accumulator into q/k/v ws -------
__device__ __forceinline__ void store_qkv_tile(const v8f& acc, int m0, int n0, int lane,
                                               const float* __restrict__ bias,
                                               __bf16* __restrict__ q_ws,
                                               __bf16* __restrict__ k_ws,
                                               __bf16* __restrict__ v_ws) {
  const int n   = n0 + (lane & 15);
  const float bv = bias[n];
  const int sec = n >> 10;          // 0=q, 1=k, 2=v
  const int j   = n & 1023;
  const int h   = j >> 6;
  const int jd  = j & 63;
#pragma unroll
  for (int v = 0; v < 8; ++v) {
    int m  = m0 + v + (lane >> 4) * 8;
    int bi = m >> 10;
    int s  = m & 1023;
    float val = acc[v] + bv;
    if (sec == 0) {
      q_ws[((bi * HEADS + h) * SEQ + s) * HDIM + jd] = (__bf16)(val * 0.125f);
    } else if (sec == 1) {
      k_ws[((bi * HEADS + h) * SEQ + s) * HDIM + jd] = (__bf16)val;
    } else {
      v_ws[((bi * HEADS + h) * HDIM + jd) * SEQ + s] = (__bf16)val;  // transposed
    }
  }
}

// ---- kernel 1: fused QKV projection (2x2 register-blocked WMMA) ------------
// C[M=8192, N=3072] = Xbf[M,1024] @ Wbf^T + bias ; scatter into q/k/v ws.
// Each wave owns a 32x32 output tile: 4 WMMAs per K-step on 2 A + 2 B frags.
__global__ __launch_bounds__(256) void qkv_gemm(const __bf16* __restrict__ xb,
                                                const __bf16* __restrict__ wb,
                                                const float* __restrict__ bias,
                                                __bf16* __restrict__ q_ws,
                                                __bf16* __restrict__ k_ws,
                                                __bf16* __restrict__ v_ws) {
  const int lane = threadIdx.x & 31;
  const int w    = threadIdx.x >> 5;
  const int n0   = blockIdx.x * 256 + w * 32;   // N tile (wave: 32 cols)
  const int m0   = blockIdx.y * 32;             // M tile (wave: 32 rows)

  const __bf16* A0 = xb + (size_t)m0 * D_MODEL;
  const __bf16* A1 = xb + (size_t)(m0 + 16) * D_MODEL;
  const __bf16* B0 = wb + (size_t)n0 * D_MODEL;         // in_proj_w rows ARE Bt rows
  const __bf16* B1 = wb + (size_t)(n0 + 16) * D_MODEL;

  v8f acc00 = {}, acc01 = {}, acc10 = {}, acc11 = {};
  for (int k0 = 0; k0 < D_MODEL; k0 += 32) {
    __builtin_prefetch(A0 + k0 + 64, 0, 1);     // global_prefetch upcoming K
    v16bf a0 = load_a_frag(A0 + k0, D_MODEL, lane);
    v16bf a1 = load_a_frag(A1 + k0, D_MODEL, lane);
    v16bf b0 = load_b_frag(B0 + k0, D_MODEL, lane);
    v16bf b1 = load_b_frag(B1 + k0, D_MODEL, lane);
    acc00 = wmma_bf16(a0, b0, acc00);
    acc01 = wmma_bf16(a0, b1, acc01);
    acc10 = wmma_bf16(a1, b0, acc10);
    acc11 = wmma_bf16(a1, b1, acc11);
  }

  store_qkv_tile(acc00, m0,      n0,      lane, bias, q_ws, k_ws, v_ws);
  store_qkv_tile(acc01, m0,      n0 + 16, lane, bias, q_ws, k_ws, v_ws);
  store_qkv_tile(acc10, m0 + 16, n0,      lane, bias, q_ws, k_ws, v_ws);
  store_qkv_tile(acc11, m0 + 16, n0 + 16, lane, bias, q_ws, k_ws, v_ws);
}

// ---- kernel 2: attention core ---------------------------------------------
// One block per (b, h, 16-row q tile). Scores staged bf16 in LDS (32KB),
// softmax with lane-shuffle reductions, mean-over-heads via f32 atomics,
// then P@V through WMMA (A-frags assembled from LDS).
__global__ __launch_bounds__(256) void attention_core(const __bf16* __restrict__ q_ws,
                                                      const __bf16* __restrict__ k_ws,
                                                      const __bf16* __restrict__ v_ws,
                                                      const float* __restrict__ time_decay,
                                                      float* __restrict__ out_w,   // [B,S,S]
                                                      __bf16* __restrict__ attn_bf) {
  __shared__ __bf16 sc[16][SEQ];      // 32 KB score/prob tile
  __shared__ float  psum[8][8][32];   // 8 KB partial PV accumulators

  const int lane = threadIdx.x & 31;
  const int w    = threadIdx.x >> 5;
  const int q0   = blockIdx.x * 16;
  const int h    = blockIdx.y;
  const int b    = blockIdx.z;
  const float td = fabsf(time_decay[0]);

  const __bf16* Qh = q_ws + ((b * HEADS + h) * SEQ + q0) * HDIM;  // 16x64, lda 64
  const __bf16* Kh = k_ws + ((b * HEADS + h) * SEQ) * HDIM;       // Bt rows=seq, ld 64
  const __bf16* Vh = v_ws + ((b * HEADS + h) * HDIM) * SEQ;       // Bt rows=hd, ld 1024

  // -- phase 1: scores = (q*scale)@k^T + bias --------------------------------
  // Q fragments are invariant across all 64 key tiles: load once.
  const v16bf qa0 = load_a_frag(Qh,      HDIM, lane);
  const v16bf qa1 = load_a_frag(Qh + 32, HDIM, lane);
  for (int t = w; t < SEQ / 16; t += 8) {
    const __bf16* Kt = Kh + t * 16 * HDIM;
    v16bf b0 = load_b_frag(Kt,      HDIM, lane);
    v16bf b1 = load_b_frag(Kt + 32, HDIM, lane);
    v8f acc = {};
    acc = wmma_bf16(qa0, b0, acc);
    acc = wmma_bf16(qa1, b1, acc);
    int j = t * 16 + (lane & 15);
#pragma unroll
    for (int v = 0; v < 8; ++v) {
      int m = v + (lane >> 4) * 8;
      int i = q0 + m;
      float sv = acc[v] - td * fabsf((float)(i - j));
      sc[m][j] = (__bf16)sv;
    }
  }
  __syncthreads();

  // -- phase 2: row softmax + mean-over-heads output -------------------------
  {
    const int row = w * 2 + (lane >> 4);   // 8 waves x 2 rows = 16 rows
    const int c0  = lane & 15;             // 16 lanes per row, stride-16 cols
    float mx = -1e30f;
#pragma unroll
    for (int jj = 0; jj < 64; ++jj)
      mx = fmaxf(mx, (float)sc[row][c0 + jj * 16]);
#pragma unroll
    for (int msk = 1; msk < 16; msk <<= 1)
      mx = fmaxf(mx, __shfl_xor(mx, msk));

    float sum = 0.0f;
#pragma unroll
    for (int jj = 0; jj < 64; ++jj) {
      int c = c0 + jj * 16;
      float e = __expf((float)sc[row][c] - mx);
      sum += e;
      sc[row][c] = (__bf16)e;
    }
#pragma unroll
    for (int msk = 1; msk < 16; msk <<= 1)
      sum += __shfl_xor(sum, msk);

    const float inv = 1.0f / sum;
    float* o2 = out_w + (size_t)((b * SEQ + q0 + row)) * SEQ;
#pragma unroll
    for (int jj = 0; jj < 64; ++jj) {
      int c = c0 + jj * 16;
      float p = (float)sc[row][c] * inv;
      sc[row][c] = (__bf16)p;
      atomicAdd(&o2[c], p * (1.0f / (float)HEADS));
    }
  }
  __syncthreads();

  // -- phase 3: attn = P @ V (16 x 64), 4 n-tiles x 2 K-halves ---------------
  {
    const int nt    = w & 3;
    const int khalf = w >> 2;
    v8f acc = {};
    const int mrow = lane & 15;
    const int hh   = lane >> 4;
    for (int ks = 0; ks < 16; ++ks) {
      int k0 = khalf * 512 + ks * 32;
      v16bf a;
      const int base = hh * 8 + k0;
#pragma unroll
      for (int v = 0; v < 4; ++v) {
        a[2 * v]     = sc[mrow][base + 2 * v];
        a[2 * v + 1] = sc[mrow][base + 2 * v + 1];
        a[8 + 2 * v] = sc[mrow][base + 16 + 2 * v];
        a[9 + 2 * v] = sc[mrow][base + 16 + 2 * v + 1];
      }
      v16bf bb = load_b_frag(Vh + nt * 16 * SEQ + k0, SEQ, lane);
      acc = wmma_bf16(a, bb, acc);
    }
#pragma unroll
    for (int v = 0; v < 8; ++v) psum[w][v][lane] = acc[v];
  }
  __syncthreads();

  if (w < 4) {
#pragma unroll
    for (int v = 0; v < 8; ++v) {
      float r = psum[w][v][lane] + psum[w + 4][v][lane];
      int m = v + (lane >> 4) * 8;
      int s = q0 + m;
      int dcol = h * HDIM + w * 16 + (lane & 15);
      attn_bf[(size_t)(b * SEQ + s) * D_MODEL + dcol] = (__bf16)r;
    }
  }
}

// ---- kernel 3: out-proj GEMM + residual + LayerNorm (fused, 1x4 blocked) ---
__global__ __launch_bounds__(256) void outproj_ln(const __bf16* __restrict__ attn_bf,
                                                  const __bf16* __restrict__ wo,   // Bt
                                                  const float* __restrict__ ob,
                                                  const float* __restrict__ x,
                                                  const float* __restrict__ ln_w,
                                                  const float* __restrict__ ln_b,
                                                  float* __restrict__ out) {
  __shared__ __bf16 yl[16][D_MODEL];  // 32 KB residual tile
  const int lane = threadIdx.x & 31;
  const int w    = threadIdx.x >> 5;
  const int m0   = blockIdx.x * 16;

  const __bf16* A = attn_bf + (size_t)m0 * D_MODEL;
  for (int g = 0; g < 2; ++g) {
    const int nb = w * 128 + g * 64;  // this pass covers 4 n-tiles
    v8f acc[4] = {v8f{}, v8f{}, v8f{}, v8f{}};
    for (int k0 = 0; k0 < D_MODEL; k0 += 32) {
      __builtin_prefetch(A + k0 + 64, 0, 1);
      v16bf a = load_a_frag(A + k0, D_MODEL, lane);  // shared by 4 WMMAs
#pragma unroll
      for (int i = 0; i < 4; ++i) {
        v16bf bb = load_b_frag(wo + (size_t)(nb + i * 16) * D_MODEL + k0, D_MODEL, lane);
        acc[i] = wmma_bf16(a, bb, acc[i]);
      }
    }
#pragma unroll
    for (int i = 0; i < 4; ++i) {
      const int n   = nb + i * 16 + (lane & 15);
      const float bv = ob[n];
#pragma unroll
      for (int v = 0; v < 8; ++v) {
        int m = v + (lane >> 4) * 8;
        float yv = acc[i][v] + bv + x[(size_t)(m0 + m) * D_MODEL + n];
        yl[m][n] = (__bf16)yv;
      }
    }
  }
  __syncthreads();

  // LayerNorm over the 16 staged rows
  const int row = w * 2 + (lane >> 4);
  const int c0  = lane & 15;
  float sum = 0.0f;
#pragma unroll
  for (int jj = 0; jj < 64; ++jj) sum += (float)yl[row][c0 + jj * 16];
#pragma unroll
  for (int msk = 1; msk < 16; msk <<= 1) sum += __shfl_xor(sum, msk);
  const float mu = sum * (1.0f / (float)D_MODEL);

  float vs = 0.0f;
#pragma unroll
  for (int jj = 0; jj < 64; ++jj) {
    float d = (float)yl[row][c0 + jj * 16] - mu;
    vs += d * d;
  }
#pragma unroll
  for (int msk = 1; msk < 16; msk <<= 1) vs += __shfl_xor(vs, msk);
  const float rstd = rsqrtf(vs * (1.0f / (float)D_MODEL) + 1e-5f);

  float* orow = out + (size_t)(m0 + row) * D_MODEL;
#pragma unroll
  for (int jj = 0; jj < 64; ++jj) {
    int c = c0 + jj * 16;
    float yv = (float)yl[row][c];
    orow[c] = (yv - mu) * rstd * ln_w[c] + ln_b[c];
  }
}

// ---------------------------------------------------------------------------
extern "C" void kernel_launch(void* const* d_in, const int* in_sizes, int n_in,
                              void* d_out, int out_size, void* d_ws, size_t ws_size,
                              hipStream_t stream) {
  const float* x  = (const float*)d_in[0];
  const float* td = (const float*)d_in[1];
  const float* wi = (const float*)d_in[2];
  const float* bi = (const float*)d_in[3];
  const float* wo = (const float*)d_in[4];
  const float* bo = (const float*)d_in[5];
  const float* lw = (const float*)d_in[6];
  const float* lb = (const float*)d_in[7];

  float* out  = (float*)d_out;                          // [B,S,d]
  float* out2 = out + (size_t)BATCH * SEQ * D_MODEL;    // [B,S,S]

  const size_t NX  = (size_t)BATCH * SEQ * D_MODEL;     // 8M
  const size_t NWI = (size_t)3 * D_MODEL * D_MODEL;     // 3M
  const size_t NWO = (size_t)D_MODEL * D_MODEL;         // 1M

  // bf16 workspace layout (bytes): x | in_w | out_w | q | k | v | attn
  char* wsb = (char*)d_ws;
  __bf16* xb   = (__bf16*)(wsb);
  __bf16* wib  = (__bf16*)(wsb + 2 * NX);
  __bf16* wob  = (__bf16*)(wsb + 2 * (NX + NWI));
  __bf16* q_ws = (__bf16*)(wsb + 2 * (NX + NWI + NWO));
  __bf16* k_ws = (__bf16*)(wsb + 2 * (2 * NX + NWI + NWO));
  __bf16* v_ws = (__bf16*)(wsb + 2 * (3 * NX + NWI + NWO));
  __bf16* attn = (__bf16*)(wsb + 2 * (4 * NX + NWI + NWO));

  // precision converts
  cvt_f32_to_bf16<<<(int)((NX + 255) / 256), 256, 0, stream>>>(x, xb, (int)NX);
  cvt_f32_to_bf16<<<(int)((NWI + 255) / 256), 256, 0, stream>>>(wi, wib, (int)NWI);
  cvt_f32_to_bf16<<<(int)((NWO + 255) / 256), 256, 0, stream>>>(wo, wob, (int)NWO);

  // zero attn-weights output (heads accumulate via atomics each call)
  const size_t NW2 = (size_t)BATCH * SEQ * SEQ;
  zero_f32<<<(int)((NW2 + 255) / 256), 256, 0, stream>>>(out2, (int)NW2);

  // fused QKV projection: grid (N/256, M/32); wave = 32x32 output tile
  qkv_gemm<<<dim3(3 * D_MODEL / 256, BATCH * SEQ / 32), 256, 0, stream>>>(
      xb, wib, bi, q_ws, k_ws, v_ws);

  // attention: grid (S/16, H, B)
  attention_core<<<dim3(SEQ / 16, HEADS, BATCH), 256, 0, stream>>>(
      q_ws, k_ws, v_ws, td, out2, attn);

  // out-proj + residual + LayerNorm: grid (M/16)
  outproj_ln<<<dim3(BATCH * SEQ / 16), 256, 0, stream>>>(
      attn, wob, bo, x, lw, lb, out);
}
